// MambaBlock_71829033059118
// MI455X (gfx1250) — compile-verified, matrix operand order checked by reading, not run on hardware
//
#include <hip/hip_runtime.h>
#include <hip/hip_bf16.h>

typedef __attribute__((ext_vector_type(16))) _Float16 v16h;
typedef __attribute__((ext_vector_type(8)))  _Float16 v8h;
typedef __attribute__((ext_vector_type(8)))  float    v8f;

static constexpr int DMODEL  = 1024;
static constexpr int DSTATE  = 16;
static constexpr int DINNER  = 2048;
static constexpr int DTRANK  = 64;
static constexpr int BB      = 4;
static constexpr int LL      = 2048;
static constexpr int MR      = BB * LL;             // 8192 rows
static constexpr int XPN     = DTRANK + 2 * DSTATE; // 96

union AFrag { v16h v; v8h h[2]; };

// ---------------- LayerNorm -> fp16 ----------------
__global__ void k_layernorm(const float* __restrict__ x, const float* __restrict__ g,
                            const float* __restrict__ b, _Float16* __restrict__ out) {
  const int row = blockIdx.x;
  const int tid = threadIdx.x;
  const float* xr = x + (size_t)row * DMODEL;
  __shared__ float red[256];

  float v[4];
  float s = 0.f;
#pragma unroll
  for (int i = 0; i < 4; ++i) { v[i] = xr[tid + i * 256]; s += v[i]; }
  red[tid] = s; __syncthreads();
  for (int st = 128; st > 0; st >>= 1) { if (tid < st) red[tid] += red[tid + st]; __syncthreads(); }
  const float mean = red[0] * (1.f / DMODEL);
  __syncthreads();

  float sq = 0.f;
#pragma unroll
  for (int i = 0; i < 4; ++i) { float d = v[i] - mean; sq += d * d; }
  red[tid] = sq; __syncthreads();
  for (int st = 128; st > 0; st >>= 1) { if (tid < st) red[tid] += red[tid + st]; __syncthreads(); }
  const float rstd = rsqrtf(red[0] * (1.f / DMODEL) + 1e-5f);

#pragma unroll
  for (int i = 0; i < 4; ++i) {
    int c = tid + i * 256;
    out[(size_t)row * DMODEL + c] = (_Float16)((v[i] - mean) * rstd * g[c] + b[c]);
  }
}

// -------- Pack W [K x N] fp32 -> WMMA B-fragment tiles (32x16), fp16 --------
// tile index = nt * (K/32) + kt ; within tile: lane*16 halves, element j maps to
// K = kt*32 + (j<8 ? 8*hi + j : 16 + 8*hi + (j-8)),  N = nt*16 + (lane&15)
__global__ void k_pack_w(const float* __restrict__ W, _Float16* __restrict__ P,
                         int K, int N) {
  const int t = blockIdx.x * blockDim.x + threadIdx.x;
  const int ktiles = K >> 5;
  const int lane = t & 31;
  const int tile = t >> 5;
  if (tile >= (N >> 4) * ktiles) return;
  const int kt = tile % ktiles;
  const int nt = tile / ktiles;
  const int hi = lane >> 4;
  const int n  = nt * 16 + (lane & 15);
  _Float16* dst = P + (size_t)tile * 512 + lane * 16;
#pragma unroll
  for (int j = 0; j < 16; ++j) {
    int kk = (j < 8) ? (hi * 8 + j) : (16 + hi * 8 + (j - 8));
    dst[j] = (_Float16)W[(size_t)(kt * 32 + kk) * N + n];
  }
}

// ---------------- WMMA GEMM: C[MxN] = A[MxK](f16,row-major) * Bpack + Res ----------------
// one wave per 32x64 output block: 2 M-tiles x 4 N-tiles of 16x16.
// All 8 waves of a block share the same 64-column strip -> the 4 packed B tiles
// per K-step are staged once per block into LDS with async global->LDS copies,
// double buffered and overlapped with the WMMA work of the previous K-step.
__global__ void k_wmma_gemm(const _Float16* __restrict__ A, const _Float16* __restrict__ Bp,
                            const float* __restrict__ Res, float* __restrict__ C,
                            int M, int N, int K) {
  __shared__ alignas(32) _Float16 bsm[2][4 * 512];   // 2 x 4KB double buffer

  const int tid  = threadIdx.x;
  const int wid  = (blockIdx.x * blockDim.x + tid) >> 5;
  const int lane = tid & 31;
  const int mtiles = M >> 5;                 // multiple of 8 -> block shares nt
  const int mt = wid % mtiles;
  const int nt = wid / mtiles;

  const int m0 = mt * 32;
  const int n0 = nt * 64;
  const int nt16 = nt * 4;
  const int hi = lane >> 4;
  const int lm = lane & 15;
  const int ktiles = K >> 5;

  // cooperative stage: 256 threads x 16B = 4KB = four 32x16 B tiles
  const int sj = tid >> 6;                   // tile 0..3
  const int so = (tid & 63) * 8;             // halves within tile
  const _Float16* bsrc = Bp + (size_t)(nt16 + sj) * ktiles * 512 + so;
  _Float16* bdst0 = &bsm[0][sj * 512 + so];
  _Float16* bdst1 = &bsm[1][sj * 512 + so];

  auto stage = [&](int kt, int s) {
    const _Float16* src = bsrc + (size_t)kt * 512;
    unsigned ldsoff = (unsigned)(unsigned long long)(s ? bdst1 : bdst0);
    asm volatile("global_load_async_to_lds_b128 %0, %1, off"
                 :: "v"(ldsoff), "v"(src) : "memory");
  };

  v8f acc[2][4] = {};

  stage(0, 0);
  asm volatile("s_wait_asynccnt 0x0" ::: "memory");
  __syncthreads();

  for (int kt = 0; kt < ktiles; ++kt) {
    const int s = kt & 1;
    if (kt + 1 < ktiles) stage(kt + 1, s ^ 1);

    // Load all four B fragments and pin them in registers: the empty asm
    // "writes" b0..b3, forcing the 8 ds_load_b128 to complete (one combined
    // s_wait_dscnt) before the 8 WMMAs, which then issue back-to-back.
    const _Float16* bb = &bsm[s][lane * 16];
    v16h b0 = *(const v16h*)(bb + 0 * 512);
    v16h b1 = *(const v16h*)(bb + 1 * 512);
    v16h b2 = *(const v16h*)(bb + 2 * 512);
    v16h b3 = *(const v16h*)(bb + 3 * 512);
    asm volatile("" : "+v"(b0), "+v"(b1), "+v"(b2), "+v"(b3));

    AFrag a[2];
#pragma unroll
    for (int i = 0; i < 2; ++i) {
      const _Float16* ap = A + (size_t)(m0 + i * 16 + lm) * K + kt * 32 + hi * 8;
      a[i].h[0] = *(const v8h*)ap;
      a[i].h[1] = *(const v8h*)(ap + 16);
    }

#pragma unroll
    for (int i = 0; i < 2; ++i) {
      acc[i][0] = __builtin_amdgcn_wmma_f32_16x16x32_f16(
          false, a[i].v, false, b0, (short)0, acc[i][0], false, false);
      acc[i][1] = __builtin_amdgcn_wmma_f32_16x16x32_f16(
          false, a[i].v, false, b1, (short)0, acc[i][1], false, false);
      acc[i][2] = __builtin_amdgcn_wmma_f32_16x16x32_f16(
          false, a[i].v, false, b2, (short)0, acc[i][2], false, false);
      acc[i][3] = __builtin_amdgcn_wmma_f32_16x16x32_f16(
          false, a[i].v, false, b3, (short)0, acc[i][3], false, false);
    }

    asm volatile("s_wait_asynccnt 0x0" ::: "memory");
    __syncthreads();   // readers of bsm[s] done; next iter may overwrite it
  }

#pragma unroll
  for (int i = 0; i < 2; ++i)
#pragma unroll
    for (int j = 0; j < 4; ++j)
#pragma unroll
      for (int r = 0; r < 8; ++r) {
        int row = m0 + i * 16 + hi * 8 + r;
        int col = n0 + j * 16 + lm;
        float val = acc[i][j][r];
        if (Res) val += Res[(size_t)row * N + col];
        C[(size_t)row * N + col] = val;
      }
}

// ---------------- depthwise causal conv (width 4) + SiLU ----------------
__global__ void k_conv_silu(const float* __restrict__ xz, const float* __restrict__ cw,
                            const float* __restrict__ cb, float* __restrict__ xc,
                            _Float16* __restrict__ xch) {
  const int idx = blockIdx.x * blockDim.x + threadIdx.x;   // MR*DINNER exact grid
  const int d   = idx & (DINNER - 1);
  const int row = idx >> 11;
  const int t   = row & (LL - 1);
  float acc = cb[d];
#pragma unroll
  for (int k = 0; k < 4; ++k) {
    int tt = t + k - 3;
    if (tt >= 0) acc += cw[d * 4 + k] * xz[(size_t)(row + k - 3) * (2 * DINNER) + d];
  }
  float s = acc / (1.f + __expf(-acc));
  xc[idx]  = s;
  xch[idx] = (_Float16)s;
}

// ---------------- x_dbl = xc @ W_xp  [MR x 96] ----------------
__global__ void k_xdbl(const float* __restrict__ xc, const float* __restrict__ Wxp,
                       float* __restrict__ xdbl) {
  const int row = blockIdx.x;
  __shared__ float srow[DINNER];
  for (int i = threadIdx.x; i < DINNER; i += blockDim.x)
    srow[i] = xc[(size_t)row * DINNER + i];
  __syncthreads();
  if (threadIdx.x < XPN) {
    float acc = 0.f;
    for (int k = 0; k < DINNER; ++k)
      acc += srow[k] * Wxp[(size_t)k * XPN + threadIdx.x];
    xdbl[(size_t)row * XPN + threadIdx.x] = acc;
  }
}

// ---------------- delta = softplus(dt_lo @ W_dt + dt_bias) ----------------
__global__ void k_delta(const float* __restrict__ xdbl, const float* __restrict__ Wdt,
                        const float* __restrict__ dtb, float* __restrict__ delta) {
  const int idx = blockIdx.x * blockDim.x + threadIdx.x;   // MR*DINNER exact grid
  const int d   = idx & (DINNER - 1);
  const int row = idx >> 11;
  const float* xr = xdbl + (size_t)row * XPN;
  float acc = dtb[d];
#pragma unroll 8
  for (int k = 0; k < DTRANK; ++k)
    acc += xr[k] * Wdt[(size_t)k * DINNER + d];
  delta[idx] = (acc > 20.f) ? acc : log1pf(__expf(acc));
}

// ---------------- selective scan, fused D-skip + SiLU(z) gate ----------------
__global__ void k_scan(const float* __restrict__ delta, const float* __restrict__ xc,
                       const float* __restrict__ xdbl, const float* __restrict__ xz,
                       const float* __restrict__ A_log, const float* __restrict__ Dskip,
                       _Float16* __restrict__ yh) {
  const int gid = blockIdx.x * blockDim.x + threadIdx.x;   // BB*DINNER threads
  const int b = gid >> 11;
  const int d = gid & (DINNER - 1);

  float A[DSTATE], h[DSTATE];
#pragma unroll
  for (int n = 0; n < DSTATE; ++n) {
    A[n] = -__expf(A_log[d * DSTATE + n]);
    h[n] = 0.f;
  }
  const float Dv = Dskip[d];

  for (int t = 0; t < LL; ++t) {
    const size_t row = (size_t)b * LL + t;
    const float dt = delta[row * DINNER + d];
    const float u  = xc[row * DINNER + d];
    const float du = dt * u;
    const float* bc = xdbl + row * XPN + DTRANK;   // [B(16) | C(16)]
    float y = 0.f;
#pragma unroll
    for (int n = 0; n < DSTATE; ++n) {
      h[n] = h[n] * __expf(dt * A[n]) + du * bc[n];
      y += h[n] * bc[DSTATE + n];
    }
    const float z = xz[row * (2 * DINNER) + DINNER + d];
    const float gate = z / (1.f + __expf(-z));
    yh[row * DINNER + d] = (_Float16)((y + u * Dv) * gate);
  }
}

extern "C" void kernel_launch(void* const* d_in, const int* in_sizes, int n_in,
                              void* d_out, int out_size, void* d_ws, size_t ws_size,
                              hipStream_t stream) {
  const float* x       = (const float*)d_in[0];
  const float* ln_g    = (const float*)d_in[1];
  const float* ln_b    = (const float*)d_in[2];
  const float* W_in    = (const float*)d_in[3];
  const float* conv_w  = (const float*)d_in[4];
  const float* conv_b  = (const float*)d_in[5];
  const float* W_xp    = (const float*)d_in[6];
  const float* W_dt    = (const float*)d_in[7];
  const float* dt_bias = (const float*)d_in[8];
  const float* A_log   = (const float*)d_in[9];
  const float* D_skip  = (const float*)d_in[10];
  const float* W_out   = (const float*)d_in[11];
  float* out = (float*)d_out;

  char* ws = (char*)d_ws;
  size_t off = 0;
  auto take = [&](size_t bytes) -> char* {
    char* p = ws + off;
    off += (bytes + 255) & ~(size_t)255;
    return p;
  };

  _Float16* xn_h   = (_Float16*)take((size_t)MR * DMODEL * 2);
  _Float16* win_p  = (_Float16*)take((size_t)DMODEL * 2 * DINNER * 2 * 2); // 1024x4096 f16
  _Float16* wout_p = (_Float16*)take((size_t)DINNER * DMODEL * 2);
  float*    xz     = (float*)   take((size_t)MR * 2 * DINNER * 4);
  float*    xc     = (float*)   take((size_t)MR * DINNER * 4);
  _Float16* xc_h   = (_Float16*)take((size_t)MR * DINNER * 2);
  float*    xdbl   = (float*)   take((size_t)MR * XPN * 4);
  float*    delta  = (float*)   take((size_t)MR * DINNER * 4);
  _Float16* yg_h   = (_Float16*)take((size_t)MR * DINNER * 2);
  (void)ws_size; (void)in_sizes; (void)n_in; (void)out_size;

  // 1) LayerNorm -> fp16
  k_layernorm<<<MR, 256, 0, stream>>>(x, ln_g, ln_b, xn_h);

  // 2) pack weights into WMMA B-fragment layout
  {
    int tiles = (4096 / 16) * (1024 / 32);            // 8192
    k_pack_w<<<tiles * 32 / 256, 256, 0, stream>>>(W_in, win_p, 1024, 4096);
  }
  {
    int tiles = (1024 / 16) * (2048 / 32);            // 4096
    k_pack_w<<<tiles * 32 / 256, 256, 0, stream>>>(W_out, wout_p, 2048, 1024);
  }

  // 3) xz = xn @ W_in  [8192 x 4096], K=1024
  {
    int waves = (MR / 32) * (4096 / 64);              // 16384
    k_wmma_gemm<<<waves / 8, 256, 0, stream>>>(xn_h, win_p, nullptr, xz, MR, 4096, 1024);
  }

  // 4) depthwise causal conv + SiLU
  k_conv_silu<<<(MR * DINNER) / 256, 256, 0, stream>>>(xz, conv_w, conv_b, xc, xc_h);

  // 5) x_dbl = xc @ W_xp
  k_xdbl<<<MR, 128, 0, stream>>>(xc, W_xp, xdbl);

  // 6) delta = softplus(dt_lo @ W_dt + dt_bias)
  k_delta<<<(MR * DINNER) / 256, 256, 0, stream>>>(xdbl, W_dt, dt_bias, delta);

  // 7) selective scan + gate -> fp16 y
  k_scan<<<(BB * DINNER) / 256, 256, 0, stream>>>(delta, xc, xdbl, xz, A_log, D_skip, yg_h);

  // 8) out = y @ W_out + x  [8192 x 1024], K=2048
  {
    int waves = (MR / 32) * (1024 / 64);              // 4096
    k_wmma_gemm<<<waves / 8, 256, 0, stream>>>(yg_h, wout_p, x, out, MR, 1024, 2048);
  }
}
